// ProteinPointCoding_86792699117898
// MI455X (gfx1250) — compile-verified, matrix operand order checked by reading, not run on hardware
//
#include <hip/hip_runtime.h>
#include <hip/hip_bf16.h>
#include <stdint.h>

typedef __attribute__((ext_vector_type(16))) __bf16 v16bf;
typedef __attribute__((ext_vector_type(8)))  float  v8f;

#define BB   32
#define NN   1024
#define HID  512
#define OUTD 256
#define SLOPE 0.2f

__device__ __forceinline__ float leaky(float v) { return v > 0.0f ? v : SLOPE * v; }

// ---------------------------------------------------------------------------
// ids from last column of xp (B, N, N+1)
// ---------------------------------------------------------------------------
__global__ void prep_ids_kernel(const float* __restrict__ xp, int* __restrict__ ids) {
    int t = blockIdx.x * blockDim.x + threadIdx.x;
    if (t >= BB * NN) return;
    int b = t >> 10, n = t & 1023;
    ids[t] = (int)xp[(size_t)b * NN * (NN + 1) + (size_t)n * (NN + 1) + NN];
}

// ---------------------------------------------------------------------------
// adjacency pack: drop id column, f32 -> bf16 (read 134MB once, write 64MB;
// the bf16 copy then fits in the 192MB L2 across both spmms)
// ---------------------------------------------------------------------------
__global__ void adj_pack_kernel(const float* __restrict__ xp, __bf16* __restrict__ Abf) {
    int t = blockIdx.x * blockDim.x + threadIdx.x;           // group of 4
    if (t >= BB * NN * NN / 4) return;
    int b = t / (NN * NN / 4);
    int r = t % (NN * NN / 4);
    int n = r >> 8;
    int j = (r & 255) << 2;
    const float* s = xp + (size_t)b * NN * (NN + 1) + (size_t)n * (NN + 1) + j;
    __bf16* d = Abf + ((size_t)b * NN + n) * NN + j;
    d[0] = (__bf16)s[0]; d[1] = (__bf16)s[1]; d[2] = (__bf16)s[2]; d[3] = (__bf16)s[3];
}

// ---------------------------------------------------------------------------
// E2[v][d] = b1[d] + sum_h emb[v][h]*W1[h][d], stored bf16 (row-major, gathered)
// ---------------------------------------------------------------------------
__global__ void e2_kernel(const float* __restrict__ emb, const float* __restrict__ W1,
                          const float* __restrict__ b1, __bf16* __restrict__ E2) {
    int t = blockIdx.x * blockDim.x + threadIdx.x;
    if (t >= 23 * HID) return;
    int v = t >> 9, d = t & 511;
    float s = b1[d];
    for (int h = 0; h < HID; ++h)
        s += emb[v * HID + h] * W1[(size_t)h * HID + d];
    E2[t] = (__bf16)s;
}

// ---------------------------------------------------------------------------
// W2T[o][h] = bf16(W2[h][o])  (pre-transposed so gemm2 B tiles are straight copies)
// ---------------------------------------------------------------------------
__global__ void w2t_kernel(const float* __restrict__ W2, __bf16* __restrict__ W2T) {
    int t = blockIdx.x * blockDim.x + threadIdx.x;
    if (t >= HID * OUTD) return;
    int o = t & 255, h = t >> 8;
    W2T[(size_t)o * HID + h] = (__bf16)W2[(size_t)h * OUTD + o];
}

// ---------------------------------------------------------------------------
// WMMA GEMM, all-bf16 operands:
//   C[M,N] = act( A[M,K] * B[K,N] (+bias) )
//   256 thr = 8 waves; block tile 64(M)x128(N); wave 2x2 wmma 16x16x32 tiles
//   A tile  : async global->LDS b128 copy (1 chunk/thread)
//   B tile  : !GATHER -> B given transposed (BT[n][k]) -> async copy (2 chunks/thr)
//             GATHER  -> rows E2[ids[k]] loaded + transposed into LDS via VGPRs
//   double-buffered LDS, pipelined with s_wait_asynccnt
//   LDS rows padded to 40 halfs (80B = 5x16B: async chunks stay 16B aligned)
// ---------------------------------------------------------------------------
template <bool GATHER, bool LEAKY, bool BIAS, bool TROUT, bool OBF16>
__global__ __launch_bounds__(256)
void gemm_wmma_kernel(const __bf16* __restrict__ A, size_t aRowStride, size_t aBatchStride,
                      const __bf16* __restrict__ Bm, size_t ldb, size_t bBatchStride,
                      const int* __restrict__ gidx, size_t gBatchStride,
                      const float* __restrict__ bias,
                      void* __restrict__ Cout, size_t ldc, size_t cBatchStride,
                      int Kdim) {
    extern __shared__ char smem[];
    constexpr int      ROWP     = 40;                       // padded halfs per LDS row
    constexpr unsigned SA_BYTES = 64u  * ROWP * 2;          // 5120
    constexpr unsigned SB_BYTES = 128u * ROWP * 2;          // 10240
    constexpr unsigned BUF      = SA_BYTES + SB_BYTES;      // 15360 (x2 buffers)

    const int tid   = threadIdx.x;
    const int lane  = tid & 31;
    const int wave  = tid >> 5;
    const int waveM = wave >> 2;
    const int waveN = wave & 3;
    const int h     = lane >> 4;
    const int lm    = lane & 15;

    const int batch   = blockIdx.z;
    const int rowBase = blockIdx.y * 64;
    const int colBase = blockIdx.x * 128;

    const __bf16* Ab = A + (size_t)batch * aBatchStride;
    const __bf16* Bb = Bm + (size_t)batch * bBatchStride;
    const int*    gb = GATHER ? (gidx + (size_t)batch * gBatchStride) : nullptr;

    const unsigned smemBase = (unsigned)(uintptr_t)smem;    // LDS byte offset (addr[31:0])

    // per-thread staging coordinates
    const int ar = tid >> 2;            // A row 0..63
    const int ac = tid & 3;             // A 16B chunk 0..3
    const int bk = tid >> 3;            // gather: B k-row 0..31
    const int bn = (tid & 7) << 4;      // gather: 16 consecutive n

    auto stage = [&](int k0, int p) {
        const unsigned base = smemBase + (unsigned)p * BUF;
        {   // A: one 16B chunk per thread (64 rows x 4 chunks)
            const __bf16* g = Ab + (size_t)(rowBase + ar) * aRowStride + k0 + (ac << 3);
            unsigned l = base + (unsigned)ar * (ROWP * 2) + ((unsigned)ac << 4);
            asm volatile("global_load_async_to_lds_b128 %0, %1, off"
                         :: "v"(l), "v"(g) : "memory");
        }
        if constexpr (!GATHER) {        // B: BT[n][k] rows, 2 chunks per thread
#pragma unroll
            for (int u = 0; u < 2; ++u) {
                int idx = tid * 2 + u;
                int n = idx >> 2, c = idx & 3;
                const __bf16* g = Bb + (size_t)(colBase + n) * ldb + k0 + (c << 3);
                unsigned l = base + SA_BYTES + (unsigned)n * (ROWP * 2) + ((unsigned)c << 4);
                asm volatile("global_load_async_to_lds_b128 %0, %1, off"
                             :: "v"(l), "v"(g) : "memory");
            }
        } else {                        // B: gather rows of E2, transpose via VGPRs
            __bf16* sB = (__bf16*)(smem + (size_t)p * BUF + SA_BYTES);
            int krow = gb[k0 + bk];
            const __bf16* src = Bb + (size_t)krow * ldb + colBase + bn;
#pragma unroll
            for (int j = 0; j < 16; ++j) sB[(size_t)(bn + j) * ROWP + bk] = src[j];
        }
    };

    // fragment K offsets (ISA 7.12.2, 16-bit operand 16x32 layout)
    int kofs[16];
#pragma unroll
    for (int j = 0; j < 16; ++j)
        kofs[j] = ((j < 8) ? 0 : 16) + (((j >> 1) & 3) << 1) + (j & 1) + (h << 3);

    const v8f vzero = {0.f, 0.f, 0.f, 0.f, 0.f, 0.f, 0.f, 0.f};
    v8f acc[2][2];
    acc[0][0] = vzero; acc[0][1] = vzero; acc[1][0] = vzero; acc[1][1] = vzero;

    stage(0, 0);
    int p = 0;
    for (int k0 = 0; k0 < Kdim; k0 += 32) {
        const bool hasNext = (k0 + 32) < Kdim;
        if (hasNext) stage(k0 + 32, p ^ 1);
        // wait for THIS buffer's async chunks (next stage's may stay in flight)
        if constexpr (GATHER) {
            if (hasNext) asm volatile("s_wait_asynccnt 0x1" ::: "memory");
            else         asm volatile("s_wait_asynccnt 0x0" ::: "memory");
        } else {
            if (hasNext) asm volatile("s_wait_asynccnt 0x3" ::: "memory");
            else         asm volatile("s_wait_asynccnt 0x0" ::: "memory");
        }
        __syncthreads();

        const __bf16* sA = (const __bf16*)(smem + (size_t)p * BUF);
        const __bf16* sB = (const __bf16*)(smem + (size_t)p * BUF + SA_BYTES);

        v16bf afr[2], bfr[2];
#pragma unroll
        for (int s = 0; s < 2; ++s) {
            const int m = waveM * 32 + s * 16 + lm;
            const int n = waveN * 32 + s * 16 + lm;
#pragma unroll
            for (int j = 0; j < 16; ++j) {
                afr[s][j] = sA[(size_t)m * ROWP + kofs[j]];
                bfr[s][j] = sB[(size_t)n * ROWP + kofs[j]];
            }
        }
#pragma unroll
        for (int ms = 0; ms < 2; ++ms)
#pragma unroll
            for (int ns = 0; ns < 2; ++ns)
                acc[ms][ns] = __builtin_amdgcn_wmma_f32_16x16x32_bf16(
                    false, afr[ms], false, bfr[ns], (short)0, acc[ms][ns], false, false);
        __syncthreads();    // protect buffer p before it is restaged
        p ^= 1;
    }

    // epilogue
#pragma unroll
    for (int ms = 0; ms < 2; ++ms) {
#pragma unroll
        for (int ns = 0; ns < 2; ++ns) {
            const int col = colBase + waveN * 32 + ns * 16 + lm;
            const float bv = BIAS ? bias[col] : 0.0f;
#pragma unroll
            for (int r = 0; r < 8; ++r) {
                const int row = rowBase + waveM * 32 + ms * 16 + r + h * 8;
                float v = acc[ms][ns][r] + bv;
                if (LEAKY) v = leaky(v);
                const size_t off = TROUT ? ((size_t)col * ldc + row) : ((size_t)row * ldc + col);
                if (OBF16) ((__bf16*)Cout)[(size_t)batch * cBatchStride + off] = (__bf16)v;
                else       ((float*)Cout)[(size_t)batch * cBatchStride + off] = v;
            }
        }
    }
}

// ---------------------------------------------------------------------------
// conv1d ('NCH','OIH'), padding (K-1,K-1), + leaky.  LOUT = LIN + 7
// ---------------------------------------------------------------------------
template <int CIN, int COUT, int LIN>
__global__ void conv_leaky_kernel(const float* __restrict__ x, const float* __restrict__ w,
                                  const float* __restrict__ bias, float* __restrict__ y) {
    constexpr int LOUT  = LIN + 7;
    constexpr int TOTAL = BB * COUT * LOUT;
    int t = blockIdx.x * blockDim.x + threadIdx.x;
    if (t >= TOTAL) return;
    int l = t % LOUT;
    int r = t / LOUT;
    int o = r % COUT;
    int b = r / COUT;

    float s = bias[o];
    const float* xb = x + (size_t)b * CIN * LIN;
    const float* wo = w + (size_t)o * CIN * 8;
    for (int i = 0; i < CIN; ++i) {
        const float* xi = xb + (size_t)i * LIN;
        const float* wi = wo + i * 8;
#pragma unroll
        for (int k = 0; k < 8; ++k) {
            int pp = l + k - 7;
            if (pp >= 0 && pp < LIN) s += wi[k] * xi[pp];
        }
    }
    y[t] = leaky(s);
}

__global__ void maxred_kernel(const float* __restrict__ x, float* __restrict__ mx) {
    int t = blockIdx.x * blockDim.x + threadIdx.x;
    if (t >= BB * 128) return;
    const float* p = x + (size_t)t * 1045;
    float m = p[0];
    for (int l = 1; l < 1045; ++l) m = fmaxf(m, p[l]);
    mx[t] = m;
}

__global__ void fc_kernel(const float* __restrict__ mx, const float* __restrict__ fcW,
                          const float* __restrict__ fcb, float* __restrict__ out) {
    int t = blockIdx.x * blockDim.x + threadIdx.x;
    if (t >= BB * OUTD) return;
    int o = t & 255, b = t >> 8;
    float s = fcb[o];
    for (int c = 0; c < 128; ++c) s += mx[b * 128 + c] * fcW[c * OUTD + o];
    out[t] = s;
}

// ---------------------------------------------------------------------------
extern "C" void kernel_launch(void* const* d_in, const int* in_sizes, int n_in,
                              void* d_out, int out_size, void* d_ws, size_t ws_size,
                              hipStream_t stream) {
    const float* xp  = (const float*)d_in[0];
    const float* emb = (const float*)d_in[1];
    const float* W1  = (const float*)d_in[2];
    const float* b1  = (const float*)d_in[3];
    const float* W2  = (const float*)d_in[4];
    const float* b2  = (const float*)d_in[5];
    const float* cw1 = (const float*)d_in[6];
    const float* cb1 = (const float*)d_in[7];
    const float* cw2 = (const float*)d_in[8];
    const float* cb2 = (const float*)d_in[9];
    const float* cw3 = (const float*)d_in[10];
    const float* cb3 = (const float*)d_in[11];
    const float* fcW = (const float*)d_in[12];
    const float* fcb = (const float*)d_in[13];
    float* out = (float*)d_out;

    // workspace arena (float-granular offsets; every region 16B aligned)
    float*  ws   = (float*)d_ws;
    __bf16* E2   = (__bf16*)ws;                              // 23*512 bf16   (5888 f)
    int*    ids  = (int*)(ws + 5888);                        // 32768 ints
    __bf16* Abf  = (__bf16*)(ws + 5888 + 32768);             // B*N*N bf16    (16777216 f)
    __bf16* W2T  = (__bf16*)(ws + 5888 + 32768 + 16777216);  // 256x512 bf16  (65536 f)
    __bf16* H1   = (__bf16*)(ws + 5888 + 32768 + 16777216 + 65536);          // B*N*512 bf16 (8388608 f)
    __bf16* H2pT = (__bf16*)(ws + 5888 + 32768 + 16777216 + 65536 + 8388608);// 256x(B*N) bf16 (4194304 f)
    float*  Xt   = ws + 5888 + 32768 + 16777216 + 65536 + 8388608 + 4194304; // B*256*N f32
    float*  C1   = Xt + (size_t)BB * OUTD * NN;
    float*  C2   = C1 + (size_t)BB * 32 * 1031;
    float*  C3   = C2 + (size_t)BB * 64 * 1038;
    float*  MX   = C3 + (size_t)BB * 128 * 1045;

    constexpr unsigned LDS_BYTES = 2u * (64u * 40 * 2 + 128u * 40 * 2);  // 30720

    // 0) packing passes (bf16 everywhere the tensor path touches)
    prep_ids_kernel<<<(BB * NN + 255) / 256, 256, 0, stream>>>(xp, ids);
    adj_pack_kernel<<<(BB * NN * NN / 4 + 255) / 256, 256, 0, stream>>>(xp, Abf);
    e2_kernel<<<(23 * HID + 255) / 256, 256, 0, stream>>>(emb, W1, b1, E2);
    w2t_kernel<<<(HID * OUTD + 255) / 256, 256, 0, stream>>>(W2, W2T);

    // 1) spmm1: H1 = leaky(A @ E2[ids])  (bf16 out), M=1024 N=512 K=1024, batched
    gemm_wmma_kernel<true, true, false, false, true>
        <<<dim3(HID / 128, NN / 64, BB), 256, LDS_BYTES, stream>>>(
        Abf, NN, (size_t)NN * NN,
        E2, HID, 0,
        ids, NN,
        nullptr,
        H1, HID, (size_t)NN * HID,
        NN);

    // 2) gemm2: H2pT[o][b*N+n] = (H1 @ W2 + b2)^T  (bf16, transposed store)
    gemm_wmma_kernel<false, false, true, true, true>
        <<<dim3(OUTD / 128, (BB * NN) / 64, 1), 256, LDS_BYTES, stream>>>(
        H1, HID, 0,
        W2T, HID, 0,
        nullptr, 0,
        b2,
        H2pT, (size_t)BB * NN, 0,
        HID);

    // 3) spmm2: Xt[b][d][n] = leaky(A @ H2p)  (f32, transposed into conv layout)
    gemm_wmma_kernel<false, true, false, true, false>
        <<<dim3(OUTD / 128, NN / 64, BB), 256, LDS_BYTES, stream>>>(
        Abf, NN, (size_t)NN * NN,
        H2pT, (size_t)BB * NN, (size_t)NN,     // B^T rows stride B*N; batch offsets +1024 along row
        nullptr, 0,
        nullptr,
        Xt, NN, (size_t)OUTD * NN,
        NN);

    // 4) conv stack + max + fc
    conv_leaky_kernel<256, 32, 1024><<<(BB * 32 * 1031 + 255) / 256, 256, 0, stream>>>(Xt, cw1, cb1, C1);
    conv_leaky_kernel<32, 64, 1031><<<(BB * 64 * 1038 + 255) / 256, 256, 0, stream>>>(C1, cw2, cb2, C2);
    conv_leaky_kernel<64, 128, 1038><<<(BB * 128 * 1045 + 255) / 256, 256, 0, stream>>>(C2, cw3, cb3, C3);
    maxred_kernel<<<(BB * 128 + 255) / 256, 256, 0, stream>>>(C3, MX);
    fc_kernel<<<(BB * OUTD + 255) / 256, 256, 0, stream>>>(MX, fcW, fcb, out);
}
